// GSA_60138132078622
// MI455X (gfx1250) — matrix operand order repaired and checked by reading, not compiled, not run on hardware
//
#include <hip/hip_runtime.h>
#include <hip/hip_bf16.h>
#include <math.h>

typedef __attribute__((ext_vector_type(16))) _Float16 v16h;
typedef __attribute__((ext_vector_type(8)))  _Float16 v8h;
typedef __attribute__((ext_vector_type(8)))  float    v8f;
typedef __attribute__((ext_vector_type(4)))  int      v4i;

#define RSQ  0.99999500003749973f   /* 1/sqrt(1+1e-5) */
#define MROW 3136
#define LDK1 264                    /* padded k-stride (halves), k=256 */
#define LDK2 40                     /* padded k-stride (halves), k=32  */

// ---- WMMA fragment loaders (layouts per CDNA5 ISA 7.12.2) ----------------
// A (16-bit, 16x32): lane L -> row (L&15); halves 0..7 = K kb..kb+7,
// halves 8..15 = K kb+16..kb+23, kb = kbase + 8*(L>=16)
__device__ __forceinline__ v16h frag_a(const _Float16* Wp, int ldk, int row,
                                       int kbase, int lane) {
  int hi = (lane >> 4) & 1;
  const _Float16* p = Wp + (size_t)row * ldk + kbase + hi * 8;
  union { v16h v; v8h a[2]; } u;
  u.a[0] = *(const v8h*)(p);
  u.a[1] = *(const v8h*)(p + 16);
  return u.v;
}
// B (16-bit, 32x16): lane L -> col (L&15); halves 0..7 = K kb..kb+7,
// halves 8..15 = K kb+16..kb+23, kb = kbase + 8*(L>=16)
__device__ __forceinline__ v16h frag_b(const _Float16* Xp, int ldk, int col,
                                       int kbase, int lane) {
  int hi = (lane >> 4) & 1;
  const _Float16* p = Xp + (size_t)col * ldk + kbase + hi * 8;
  union { v16h v; v8h a[2]; } u;
  u.a[0] = *(const v8h*)(p);
  u.a[1] = *(const v8h*)(p + 16);
  return u.v;
}
__device__ __forceinline__ v8f wmma_f16(v16h a, v16h b, v8f c) {
  return __builtin_amdgcn_wmma_f32_16x16x32_f16(false, a, false, b,
                                                (short)0, c, false, false);
}

// ---- async global->LDS 16B copy (ASYNCcnt path), with portable fallback --
#if __has_builtin(__builtin_amdgcn_global_load_async_to_lds_b128)
#define HAS_ASYNC_LDS 1
__device__ __forceinline__ void copy16_g2l(const _Float16* g, _Float16* l) {
  __builtin_amdgcn_global_load_async_to_lds_b128(
      (__attribute__((address_space(1))) v4i*)g,
      (__attribute__((address_space(3))) v4i*)l, 0, 0);
}
__device__ __forceinline__ void wait_async() {
  asm volatile("s_wait_asynccnt 0" ::: "memory");
}
#else
#define HAS_ASYNC_LDS 0
__device__ __forceinline__ void copy16_g2l(const _Float16* g, _Float16* l) {
  *(v8h*)l = *(const v8h*)g;
}
__device__ __forceinline__ void wait_async() {}
#endif

// ---- kernel 0: weights fp32 -> fp16 --------------------------------------
__global__ void __launch_bounds__(256)
k_cvt(const float* __restrict__ w1, const float* __restrict__ wdr,
      _Float16* __restrict__ w1h, _Float16* __restrict__ wdrh) {
  int i = blockIdx.x * 256 + threadIdx.x;
  if (i < 65536)       w1h[i]          = (_Float16)w1[i];
  else if (i < 98304)  wdrh[i - 65536] = (_Float16)wdr[i - 65536];
}

// ---- kernel 1: fused conv1+bn+relu -> dr+bias+bn+relu --------------------
// grid (49, 16), block 256 (8 waves). One 64-column tile per workgroup.
// d is written as f16 (halves HBM traffic; cov kernel consumes f16 anyway).
__global__ void __launch_bounds__(256)
k_fused(const float* __restrict__ x, const _Float16* __restrict__ w1h,
        const _Float16* __restrict__ wdrh,
        const float* __restrict__ g1,  const float* __restrict__ b1,
        const float* __restrict__ b_dr, const float* __restrict__ g_dr,
        const float* __restrict__ be_dr, _Float16* __restrict__ dbuf) {
  __shared__ _Float16 SH[64 * LDK1];     // X^T, then reused as Out1^T
  __shared__ float s1[256], bo1[256];
  __shared__ float sdr[128], bdr[128], bedr[128];
  const int b   = blockIdx.y;
  const int m0  = blockIdx.x * 64;
  const int tid = threadIdx.x;
  const int lane = tid & 31, wv = tid >> 5;
  const int hi = (lane >> 4) & 1, l15 = lane & 15;

  s1[tid]  = g1[tid] * RSQ;
  bo1[tid] = b1[tid];
  if (tid < 128) { sdr[tid] = g_dr[tid] * RSQ; bdr[tid] = b_dr[tid]; bedr[tid] = be_dr[tid]; }

  // stage X^T into LDS: SH[m_local][c]  (coalesced global reads along m)
  for (int it = 0; it < 64; ++it) {
    int mloc = tid & 63;
    int crow = (tid >> 6) + it * 4;
    float v = x[((size_t)b * 256 + crow) * MROW + m0 + mloc];
    SH[mloc * LDK1 + crow] = (_Float16)v;
  }
  __syncthreads();

  // GEMM1: out1(256x64) = W1(256x256) * X(256x64); wave -> 2 Mtiles x 4 Ntiles
  v8f acc[2][4] = {};
  for (int ks = 0; ks < 8; ++ks) {
    int kb = ks * 32;
    v16h a0 = frag_a(w1h, 256, (2 * wv)     * 16 + l15, kb, lane);
    v16h a1 = frag_a(w1h, 256, (2 * wv + 1) * 16 + l15, kb, lane);
#pragma unroll
    for (int nt = 0; nt < 4; ++nt) {
      v16h bf = frag_b(SH, LDK1, nt * 16 + l15, kb, lane);
      acc[0][nt] = wmma_f16(a0, bf, acc[0][nt]);
      acc[1][nt] = wmma_f16(a1, bf, acc[1][nt]);
    }
  }
  __syncthreads();   // done reading SH as X^T

  // bn+relu, repack transposed f16 Out1^T[n][c] into SH
  for (int mt = 0; mt < 2; ++mt) {
    int row0 = (2 * wv + mt) * 16 + hi * 8;
    for (int nt = 0; nt < 4; ++nt) {
      int n = nt * 16 + l15;
      v8h pk;
#pragma unroll
      for (int e = 0; e < 8; ++e) {
        int row = row0 + e;
        float val = acc[mt][nt][e] * s1[row] + bo1[row];
        pk[e] = (_Float16)fmaxf(val, 0.0f);
      }
      *(v8h*)(&SH[n * LDK1 + row0]) = pk;
    }
  }
  __syncthreads();

  // GEMM2: d(128x64) = Wdr(128x256) * Out1(256x64); wave -> 1 Mtile x 4 Ntiles
  v8f acc2[4] = {};
  for (int ks = 0; ks < 8; ++ks) {
    int kb = ks * 32;
    v16h a = frag_a(wdrh, 256, wv * 16 + l15, kb, lane);
#pragma unroll
    for (int nt = 0; nt < 4; ++nt) {
      v16h bf = frag_b(SH, LDK1, nt * 16 + l15, kb, lane);
      acc2[nt] = wmma_f16(a, bf, acc2[nt]);
    }
  }
  // bias + bn + relu, store d (f16)
  for (int nt = 0; nt < 4; ++nt) {
    int m = m0 + nt * 16 + l15;
#pragma unroll
    for (int e = 0; e < 8; ++e) {
      int c = wv * 16 + e + hi * 8;
      float val = (acc2[nt][e] + bdr[c]) * sdr[c] + bedr[c];
      dbuf[((size_t)b * 128 + c) * MROW + m] = (_Float16)fmaxf(val, 0.0f);
    }
  }
}

// ---- kernel 2: Gram matrix via WMMA (async-LDS staged), bn + w_row -------
// cov = (G - M*mu_i*mu_j)/M with G = d @ d^T, so the WMMA inner loop runs on
// raw f16 d -> staging is a pure byte copy -> GLOBAL_LOAD_ASYNC_TO_LDS_B128.
// grid 16 (one workgroup per batch), block 256 (8 waves).
__global__ void __launch_bounds__(256)
k_cov_r(const _Float16* __restrict__ dbuf, const float* __restrict__ g_row,
        const float* __restrict__ be_row, const float* __restrict__ w_row,
        const float* __restrict__ b_row, float* __restrict__ rbuf) {
  __shared__ _Float16 XL[128 * LDK2];    // 10 KB staging chunk
  __shared__ float pm[256];
  __shared__ float meanL[128], srow[128], berow[128];
  const int b = blockIdx.x;
  const int tid = threadIdx.x, lane = tid & 31, wv = tid >> 5;
  const int hi = (lane >> 4) & 1, l15 = lane & 15;

  // row means of d[b]  (2 threads per channel)
  {
    int c = tid >> 1, p = tid & 1;
    const _Float16* dp = dbuf + ((size_t)b * 128 + c) * MROW + p * 1568;
    float s = 0.f;
    for (int i = 0; i < 1568; ++i) s += (float)dp[i];
    pm[tid] = s;
  }
  __syncthreads();
  if (tid < 128) {
    meanL[tid] = (pm[2 * tid] + pm[2 * tid + 1]) * (1.0f / 3136.0f);
    srow[tid]  = g_row[tid] * RSQ;
    berow[tid] = be_row[tid];
  }
  __syncthreads();

  // G = d * d^T : wave wv owns row strip [16wv, 16wv+16) x all 128 cols
  v8f acc[8] = {};
  for (int ks = 0; ks < 98; ++ks) {
    int k0 = ks * 32;
    {
      // 128 rows x 64B per chunk = 512 x 16B transfers; 2 per thread
      int c  = tid >> 1;
      int q0 = (tid & 1) * 2;
      const _Float16* gp = dbuf + ((size_t)b * 128 + c) * MROW + k0;
#pragma unroll
      for (int q = q0; q < q0 + 2; ++q)
        copy16_g2l(gp + q * 8, &XL[c * LDK2 + q * 8]);
    }
    wait_async();
    __syncthreads();
    v16h a = frag_a(XL, LDK2, wv * 16 + l15, 0, lane);
#pragma unroll
    for (int nt = 0; nt < 8; ++nt) {
      v16h bf = frag_b(XL, LDK2, nt * 16 + l15, 0, lane);
      acc[nt] = wmma_f16(a, bf, acc[nt]);
    }
    __syncthreads();
  }

  // covbn + grouped conv (w_row) straight from accumulators.
  // C/D layout: lane holds col c2 = nt*16+l15; vgpr e -> row c = 16wv+e+8*hi.
  const float invM = 1.0f / 3136.0f;
  for (int e = 0; e < 8; ++e) {
    int c = wv * 16 + e + hi * 8;
    float mui = meanL[c], sc = srow[c], be = berow[c];
    float cb[8];
#pragma unroll
    for (int nt = 0; nt < 8; ++nt) {
      int c2 = nt * 16 + l15;
      cb[nt] = (acc[nt][e] * invM - mui * meanL[c2]) * sc + be;
    }
    for (int j = 0; j < 4; ++j) {
      const float* wr = w_row + ((size_t)c * 4 + j) * 128 + l15;
      float p = 0.f;
#pragma unroll
      for (int nt = 0; nt < 8; ++nt)
        p += wr[nt * 16] * cb[nt];
      // reduce over the 16 lanes of this half-wave
      p += __shfl_xor(p, 1, 32);
      p += __shfl_xor(p, 2, 32);
      p += __shfl_xor(p, 4, 32);
      p += __shfl_xor(p, 8, 32);
      if (l15 == 0)
        rbuf[(size_t)b * 512 + c * 4 + j] = p + b_row[c * 4 + j];
    }
  }
}

// ---- kernel 3: att = sigmoid(w_fc @ r + b_fc) ----------------------------
__global__ void __launch_bounds__(256)
k_att(const float* __restrict__ w_fc, const float* __restrict__ b_fc,
      const float* __restrict__ rbuf, float* __restrict__ att) {
  int b = blockIdx.x, o = threadIdx.x;
  __shared__ float rs[512];
  rs[o]       = rbuf[b * 512 + o];
  rs[o + 256] = rbuf[b * 512 + o + 256];
  __syncthreads();
  float acc = b_fc[o];
  const float* wp = w_fc + (size_t)o * 512;
  for (int k = 0; k < 512; ++k) acc += wp[k] * rs[k];
  att[b * 256 + o] = 1.0f / (1.0f + expf(-acc));
}

// ---- kernel 4: out = x * att[b,c] (float4, bandwidth-bound) --------------
__global__ void __launch_bounds__(256)
k_scale(const float* __restrict__ x, const float* __restrict__ att,
        float* __restrict__ out) {
  size_t i4 = (size_t)blockIdx.x * 256 + threadIdx.x;
  size_t f  = i4 * 4;
  int bo = (int)(f / MROW);          // = b*256 + channel (3136 % 4 == 0)
  float a = att[bo];
  const float4 xv = ((const float4*)x)[i4];
  float4 ov;
  ov.x = xv.x * a; ov.y = xv.y * a; ov.z = xv.z * a; ov.w = xv.w * a;
  ((float4*)out)[i4] = ov;
}

extern "C" void kernel_launch(void* const* d_in, const int* in_sizes, int n_in,
                              void* d_out, int out_size, void* d_ws, size_t ws_size,
                              hipStream_t stream) {
  const float* x      = (const float*)d_in[0];
  const float* w1     = (const float*)d_in[1];
  const float* g1     = (const float*)d_in[2];
  const float* b1     = (const float*)d_in[3];
  const float* wdr    = (const float*)d_in[4];
  const float* b_dr   = (const float*)d_in[5];
  const float* g_dr   = (const float*)d_in[6];
  const float* be_dr  = (const float*)d_in[7];
  const float* g_row  = (const float*)d_in[8];
  const float* be_row = (const float*)d_in[9];
  const float* w_row  = (const float*)d_in[10];
  const float* b_row  = (const float*)d_in[11];
  const float* w_fc   = (const float*)d_in[12];
  const float* b_fc   = (const float*)d_in[13];

  char* ws = (char*)d_ws;
  _Float16* w1h  = (_Float16*)(ws);                          // 128 KB
  _Float16* wdrh = (_Float16*)(ws + 131072);                 // 64 KB
  _Float16* dbuf = (_Float16*)(ws + 196608);                 // 12.85 MB (f16)
  float*    rbuf = (float*)(ws + 196608 + 12845056);         // 32 KB
  float*    attb = (float*)(ws + 196608 + 12845056 + 32768); // 16 KB
  float*    out  = (float*)d_out;

  k_cvt  <<<384, 256, 0, stream>>>(w1, wdr, w1h, wdrh);
  k_fused<<<dim3(49, 16), 256, 0, stream>>>(x, w1h, wdrh, g1, b1,
                                            b_dr, g_dr, be_dr, dbuf);
  k_cov_r<<<16, 256, 0, stream>>>(dbuf, g_row, be_row, w_row, b_row, rbuf);
  k_att  <<<16, 256, 0, stream>>>(w_fc, b_fc, rbuf, attb);
  k_scale<<<12544, 256, 0, stream>>>(x, attb, out);
}
